// FlashPairBiasAttention_29583734735205
// MI455X (gfx1250) — matrix operand order repaired and checked by reading, not compiled, not run on hardware
//
#include <hip/hip_runtime.h>
#include <math.h>

#define BDIM   2
#define NSEQ   768
#define DMODEL 512
#define NH     16
#define DH     32
#define INNER  512
#define PD     128
#define LN_EPS 1e-5f

#define KDIM      512          // K is 512 for every GEMM in this op
#define BS_STRIDE 520          // padded LDS row stride (halfs): bank step = 4 dwords
#define PB_STRIDE 136          // pair-bias LDS row stride (halfs)
#define PT_STRIDE 40           // attention P-tile LDS row stride (halfs)

typedef __attribute__((ext_vector_type(16))) __bf16 v16bf;
typedef __attribute__((ext_vector_type(8)))  __bf16 v8bf;
typedef __attribute__((ext_vector_type(8)))  float  v8f;
typedef __attribute__((ext_vector_type(4)))  float  v4f;

__device__ __forceinline__ __bf16 f2bf(float f) {
  unsigned u = __float_as_uint(f);
  u += 0x7fffu + ((u >> 16) & 1u);          // round-to-nearest-even
  unsigned short s = (unsigned short)(u >> 16);
  return __builtin_bit_cast(__bf16, s);
}

__device__ __forceinline__ v16bf cat16(v8bf lo, v8bf hi) {
  return __builtin_shufflevector(lo, hi, 0,1,2,3,4,5,6,7,8,9,10,11,12,13,14,15);
}

__device__ __forceinline__ float wave_sum32(float x) {
  #pragma unroll
  for (int off = 16; off > 0; off >>= 1) x += __shfl_xor(x, off, 32);
  return x;
}
// reductions confined to 16-lane halves (xor of bits 0..3 never crosses bit 4)
__device__ __forceinline__ float grp16_max(float x) {
  #pragma unroll
  for (int off = 8; off > 0; off >>= 1) x = fmaxf(x, __shfl_xor(x, off, 32));
  return x;
}
__device__ __forceinline__ float grp16_sum(float x) {
  #pragma unroll
  for (int off = 8; off > 0; off >>= 1) x += __shfl_xor(x, off, 32);
  return x;
}

// gfx1250 async global->LDS copy (ASYNCcnt tracked); 16B per lane
__device__ __forceinline__ void async_load_b128(unsigned lds_off, const void* gaddr) {
  asm volatile("global_load_async_to_lds_b128 %0, %1, off"
               :: "v"(lds_off), "v"((unsigned long long)(uintptr_t)gaddr)
               : "memory");
}
__device__ __forceinline__ void wait_async0() {
  asm volatile("s_wait_asynccnt 0x0" ::: "memory");
}

// ---------------- node LayerNorm -> bf16, one wave per (b,n) row ----------------
__global__ __launch_bounds__(256) void k_node_ln(const float* __restrict__ x,
                                                 const float* __restrict__ w,
                                                 const float* __restrict__ bb,
                                                 __bf16* __restrict__ xln) {
  int task = blockIdx.x * 8 + (threadIdx.x >> 5);   // b*NSEQ + n
  int lane = threadIdx.x & 31;
  const float* row = x + (size_t)task * DMODEL;
  float v[16]; float s = 0.f;
  #pragma unroll
  for (int e = 0; e < 16; e++) { v[e] = row[lane * 16 + e]; s += v[e]; }
  s = wave_sum32(s); float mean = s * (1.0f / DMODEL);
  float vs = 0.f;
  #pragma unroll
  for (int e = 0; e < 16; e++) { float d = v[e] - mean; vs += d * d; }
  vs = wave_sum32(vs);
  float rstd = rsqrtf(vs * (1.0f / DMODEL) + LN_EPS);
  __bf16* orow = xln + (size_t)task * DMODEL;
  #pragma unroll
  for (int e = 0; e < 16; e++) {
    int c = lane * 16 + e;
    orow[c] = f2bf((v[e] - mean) * rstd * w[c] + bb[c]);
  }
}

// ---------------- f32 [K][Nc] -> bf16 transposed [Nc][K] ----------------
__global__ __launch_bounds__(256) void k_transpose(const float* __restrict__ in,
                                                   __bf16* __restrict__ out,
                                                   int K, int Nc) {
  int idx = blockIdx.x * blockDim.x + threadIdx.x;
  if (idx >= K * Nc) return;
  int k = idx / Nc, n = idx % Nc;
  out[(size_t)n * K + k] = f2bf(in[idx]);
}

// ---------------- bf16 WMMA GEMM: C[M,Nc] = A[M,512] * Wt[Nc,512]^T + bias ------
// block = 8 waves = 128(M) x 64(N) tile; 64x512 weight slab staged in LDS via
// global_load_async_to_lds_b128, then fed to v_wmma_f32_16x16x32_bf16 from LDS.
__global__ __launch_bounds__(256) void k_gemm(const __bf16* __restrict__ A,
                                              const __bf16* __restrict__ Wt,
                                              const float* __restrict__ bias,
                                              float* __restrict__ C,
                                              int Nc) {
  extern __shared__ __bf16 Bs[];              // 64 rows x BS_STRIDE halfs
  int wid = threadIdx.x >> 5, lane = threadIdx.x & 31;
  int r = lane & 15, half = lane >> 4;
  int tilesN4 = Nc >> 6;
  int tm8 = blockIdx.x / tilesN4, tn4 = blockIdx.x % tilesN4;
  int col0 = tn4 * 64;

  // ---- async-stage the 64x512 slab of Wt into LDS (4096 x 16B chunks) ----
  unsigned ldsbase = (unsigned)(uintptr_t)Bs;
  #pragma unroll
  for (int i = 0; i < 16; i++) {
    int c = i * 256 + (int)threadIdx.x;
    int col = c >> 6;                         // 64 chunks (of 8 halfs) per row
    int kp  = (c & 63) * 8;
    async_load_b128(ldsbase + (unsigned)(col * BS_STRIDE + kp) * 2u,
                    Wt + (size_t)(col0 + col) * KDIM + kp);
  }
  wait_async0();
  __syncthreads();

  int tm = tm8 * 8 + wid;
  const __bf16* arow = A + (size_t)(tm * 16 + r) * KDIM;
  v8f acc[4] = {};
  v16bf a_cur = cat16(*(const v8bf*)(arow + half * 16),
                      *(const v8bf*)(arow + half * 16 + 8));
  for (int k0 = 0; k0 < KDIM; k0 += 32) {
    int k1 = (k0 + 32 < KDIM) ? (k0 + 32) : k0;     // software pipeline A loads
    v16bf a_nxt = cat16(*(const v8bf*)(arow + k1 + half * 16),
                        *(const v8bf*)(arow + k1 + half * 16 + 8));
    #pragma unroll
    for (int t = 0; t < 4; t++) {
      const __bf16* brow = Bs + (t * 16 + r) * BS_STRIDE + k0 + half * 16;
      acc[t] = __builtin_amdgcn_wmma_f32_16x16x32_bf16(
          false, a_cur, false,
          cat16(*(const v8bf*)brow, *(const v8bf*)(brow + 8)),
          (short)0, acc[t], false, false);
    }
    a_cur = a_nxt;
  }
  #pragma unroll
  for (int t = 0; t < 4; t++) {
    int col = col0 + t * 16 + r;
    float bv = bias[col];
    float* crow = C + (size_t)(tm * 16 + 8 * half) * Nc + col;
    #pragma unroll
    for (int rr = 0; rr < 8; rr++) crow[(size_t)rr * Nc] = acc[t][rr] + bv;
  }
}

// -------- q/k LayerNorm over INNER + repack head-major; v -> d-major transpose ---
__global__ __launch_bounds__(256) void k_qkv_post(const float* __restrict__ QKV,
                                                  const float* __restrict__ qw, const float* __restrict__ qb,
                                                  const float* __restrict__ kw, const float* __restrict__ kb,
                                                  __bf16* __restrict__ Qo, __bf16* __restrict__ Ko,
                                                  __bf16* __restrict__ VtO) {
  int task = blockIdx.x * 8 + (threadIdx.x >> 5);   // b*NSEQ + n
  int lane = threadIdx.x & 31;
  int b = task / NSEQ, n = task % NSEQ;
  const float* row = QKV + (size_t)task * (3 * INNER);
  #pragma unroll
  for (int seg = 0; seg < 2; seg++) {
    const float* src = row + seg * INNER;
    const float* w = seg ? kw : qw;
    const float* bbv = seg ? kb : qb;
    __bf16* dst = seg ? Ko : Qo;
    float v[16]; float s = 0.f;
    #pragma unroll
    for (int e = 0; e < 16; e++) { v[e] = src[lane * 16 + e]; s += v[e]; }
    s = wave_sum32(s); float mean = s * (1.0f / INNER);
    float vs = 0.f;
    #pragma unroll
    for (int e = 0; e < 16; e++) { float d = v[e] - mean; vs += d * d; }
    vs = wave_sum32(vs);
    float rstd = rsqrtf(vs * (1.0f / INNER) + LN_EPS);
    #pragma unroll
    for (int e = 0; e < 16; e++) {
      int c = lane * 16 + e; int h = c >> 5, d = c & 31;
      dst[((size_t)(b * NH + h) * NSEQ + n) * DH + d] =
          f2bf((v[e] - mean) * rstd * w[c] + bbv[c]);
    }
  }
  const float* src = row + 2 * INNER;
  #pragma unroll
  for (int e = 0; e < 16; e++) {
    int c = lane * 16 + e; int h = c >> 5, d = c & 31;
    VtO[((size_t)(b * NH + h) * DH + d) * NSEQ + n] = f2bf(src[c]);
  }
}

// -------- pair LN + 128->16 projection via WMMA + mask -> bias[b][h][i][j] ------
__global__ __launch_bounds__(256) void k_pair_bias(const float* __restrict__ pair,
                                                   const unsigned char* __restrict__ mask,
                                                   const float* __restrict__ pw,
                                                   const float* __restrict__ pb,
                                                   const __bf16* __restrict__ bwT,   // [NH][PD]
                                                   float* __restrict__ Bias) {
  __shared__ __bf16 pbs[8][16 * PB_STRIDE];
  int wid = threadIdx.x >> 5, lane = threadIdx.x & 31;
  int task = blockIdx.x * 8 + wid;                 // (b*NSEQ+i)*48 + jt
  int jt = task % (NSEQ / 16);
  int bi = task / (NSEQ / 16);
  int i = bi % NSEQ, b = bi / NSEQ;
  __bf16* P = pbs[wid];
  const float* base = pair + (((size_t)(b * NSEQ + i)) * NSEQ + jt * 16) * PD;
  for (int jj = 0; jj < 16; jj++) {               // LN each of 16 pair rows (PD=128)
    const float* prow = base + (size_t)jj * PD;
    v4f x = *(const v4f*)(prow + lane * 4);
    float s = x[0] + x[1] + x[2] + x[3];
    s = wave_sum32(s); float mean = s * (1.0f / PD);
    float vs = 0.f;
    #pragma unroll
    for (int c = 0; c < 4; c++) { float d = x[c] - mean; vs += d * d; }
    vs = wave_sum32(vs);
    float rstd = rsqrtf(vs * (1.0f / PD) + LN_EPS);
    #pragma unroll
    for (int c = 0; c < 4; c++) {
      int e = lane * 4 + c;
      P[jj * PB_STRIDE + e] = f2bf((x[c] - mean) * rstd * pw[e] + pb[e]);
    }
  }
  int r = lane & 15, half = lane >> 4;
  v8f acc = {};
  #pragma unroll
  for (int c = 0; c < 4; c++) {                   // K = 128 = 4 x 32
    const __bf16* ar = P + r * PB_STRIDE + c * 32;
    v16bf a = cat16(*(const v8bf*)(ar + half * 16), *(const v8bf*)(ar + half * 16 + 8));
    const __bf16* br = bwT + r * PD + c * 32 + half * 16;
    acc = __builtin_amdgcn_wmma_f32_16x16x32_bf16(
        false, a, false, cat16(*(const v8bf*)br, *(const v8bf*)(br + 8)),
        (short)0, acc, false, false);
  }
  int h = r, jbase = jt * 16 + 8 * half;          // C: rows j, cols h
  const unsigned char* mrow = mask + ((size_t)(b * NSEQ + i)) * NSEQ + jbase;
  float* brow = Bias + (((size_t)(b * NH + h) * NSEQ + i)) * NSEQ + jbase;
  #pragma unroll
  for (int rr = 0; rr < 8; rr++)
    brow[rr] = mrow[rr] ? acc[rr] : -10000.0f;
}

// -------- flash attention: one wave per (b,h, 16-row i-tile) ---------------------
__global__ __launch_bounds__(256) void k_attn(const __bf16* __restrict__ Q,
                                              const __bf16* __restrict__ Km,
                                              const __bf16* __restrict__ Vt,
                                              const float* __restrict__ Bias,
                                              const float* __restrict__ G,
                                              __bf16* __restrict__ O) {
  __shared__ __bf16 Pt[8][16 * PT_STRIDE];
  const float SCALE = 0.17677669529663687f;       // 1/sqrt(32)
  int wid = threadIdx.x >> 5, lane = threadIdx.x & 31;
  int r = lane & 15, half = lane >> 4;
  int task = blockIdx.x * 8 + wid;                // bh*48 + it
  int it = task % (NSEQ / 16), bh = task / (NSEQ / 16);
  int b = bh / NH, h = bh % NH;
  int i0 = it * 16;
  const __bf16* qrow = Q + ((size_t)bh * NSEQ + i0 + r) * DH;
  v16bf qa = cat16(*(const v8bf*)(qrow + half * 16), *(const v8bf*)(qrow + half * 16 + 8));
  const __bf16* kbase = Km + (size_t)bh * NSEQ * DH;
  const __bf16* vtb   = Vt + (size_t)bh * DH * NSEQ;
  const float* bbase  = Bias + ((size_t)bh * NSEQ + i0 + 8 * half) * NSEQ + r;
  __bf16* P = Pt[wid];
  v8f acc0 = {}, acc1 = {};
  float Mr[8], Lr[8];
  #pragma unroll
  for (int rr = 0; rr < 8; rr++) { Mr[rr] = -1e30f; Lr[rr] = 0.f; }

  for (int j0 = 0; j0 < NSEQ; j0 += 32) {
    v8f s0 = {}, s1 = {};
    const __bf16* k0p = kbase + (size_t)(j0 + r) * DH + half * 16;
    s0 = __builtin_amdgcn_wmma_f32_16x16x32_bf16(
        false, qa, false, cat16(*(const v8bf*)k0p, *(const v8bf*)(k0p + 8)),
        (short)0, s0, false, false);
    const __bf16* k1p = kbase + (size_t)(j0 + 16 + r) * DH + half * 16;
    s1 = __builtin_amdgcn_wmma_f32_16x16x32_bf16(
        false, qa, false, cat16(*(const v8bf*)k1p, *(const v8bf*)(k1p + 8)),
        (short)0, s1, false, false);

    float p0[8], p1[8];
    #pragma unroll
    for (int rr = 0; rr < 8; rr++) {              // online softmax per row
      const float* bp = bbase + (size_t)rr * NSEQ + j0;
      __builtin_prefetch(bp + 32, 0, 1);          // next j-block bias row
      float a = s0[rr] * SCALE + bp[0];
      float c = s1[rr] * SCALE + bp[16];
      float mn = fmaxf(Mr[rr], grp16_max(fmaxf(a, c)));
      float corr = __expf(Mr[rr] - mn);
      float e0 = __expf(a - mn), e1 = __expf(c - mn);
      Lr[rr] = Lr[rr] * corr + grp16_sum(e0 + e1);
      Mr[rr] = mn;
      acc0[rr] *= corr; acc1[rr] *= corr;
      p0[rr] = e0; p1[rr] = e1;
    }
    #pragma unroll
    for (int rr = 0; rr < 8; rr++) {              // stage P (16x32) via LDS
      P[(rr + 8 * half) * PT_STRIDE + r]      = f2bf(p0[rr]);
      P[(rr + 8 * half) * PT_STRIDE + 16 + r] = f2bf(p1[rr]);
    }
    const __bf16* prow = P + r * PT_STRIDE;
    v16bf pa = cat16(*(const v8bf*)(prow + half * 16), *(const v8bf*)(prow + half * 16 + 8));
    const __bf16* v0p = vtb + (size_t)r * NSEQ + j0 + half * 16;
    acc0 = __builtin_amdgcn_wmma_f32_16x16x32_bf16(
        false, pa, false, cat16(*(const v8bf*)v0p, *(const v8bf*)(v0p + 8)),
        (short)0, acc0, false, false);
    const __bf16* v1p = vtb + (size_t)(16 + r) * NSEQ + j0 + half * 16;
    acc1 = __builtin_amdgcn_wmma_f32_16x16x32_bf16(
        false, pa, false, cat16(*(const v8bf*)v1p, *(const v8bf*)(v1p + 8)),
        (short)0, acc1, false, false);
  }
  #pragma unroll
  for (int rr = 0; rr < 8; rr++) {                // normalize + sigmoid gate
    int i = i0 + rr + 8 * half;
    float inv = 1.0f / Lr[rr];
    size_t gi = ((size_t)(b * NSEQ + i)) * INNER + h * DH;
    float g0 = G[gi + r], g1 = G[gi + 16 + r];
    O[gi + r]      = f2bf(acc0[rr] * inv * (1.0f / (1.0f + __expf(-g0))));
    O[gi + 16 + r] = f2bf(acc1[rr] * inv * (1.0f / (1.0f + __expf(-g1))));
  }
}

extern "C" void kernel_launch(void* const* d_in, const int* in_sizes, int n_in,
                              void* d_out, int out_size, void* d_ws, size_t ws_size,
                              hipStream_t stream) {
  (void)in_sizes; (void)n_in; (void)out_size; (void)ws_size;
  const float* node = (const float*)d_in[0];
  const float* pair = (const float*)d_in[1];
  const unsigned char* mask = (const unsigned char*)d_in[2];
  const float* nnw  = (const float*)d_in[3];
  const float* nnb  = (const float*)d_in[4];
  const float* qkvw = (const float*)d_in[5];
  const float* qkvb = (const float*)d_in[6];
  const float* gw   = (const float*)d_in[7];
  const float* gb   = (const float*)d_in[8];
  const float* qlnw = (const float*)d_in[9];
  const float* qlnb = (const float*)d_in[10];
  const float* klnw = (const float*)d_in[11];
  const float* klnb = (const float*)d_in[12];
  const float* pnw  = (const float*)d_in[13];
  const float* pnb  = (const float*)d_in[14];
  const float* bw   = (const float*)d_in[15];
  const float* ow   = (const float*)d_in[16];
  const float* ob   = (const float*)d_in[17];

  char* base = (char*)d_ws;
  size_t off = 0;
  auto carve = [&](size_t bytes) -> char* {
    char* p = base + off;
    off = (off + bytes + 255) & ~(size_t)255;
    return p;
  };
  __bf16* xln   = (__bf16*)carve((size_t)BDIM * NSEQ * DMODEL * 2);
  __bf16* wqkvT = (__bf16*)carve((size_t)DMODEL * 3 * INNER * 2);
  __bf16* wgT   = (__bf16*)carve((size_t)DMODEL * INNER * 2);
  __bf16* woutT = (__bf16*)carve((size_t)INNER * DMODEL * 2);
  __bf16* bwT   = (__bf16*)carve((size_t)PD * NH * 2);
  float*  qkvf  = (float*)carve((size_t)BDIM * NSEQ * 3 * INNER * 4);
  float*  gf    = (float*)carve((size_t)BDIM * NSEQ * INNER * 4);
  __bf16* Qb    = (__bf16*)carve((size_t)BDIM * NH * NSEQ * DH * 2);
  __bf16* Kb    = (__bf16*)carve((size_t)BDIM * NH * NSEQ * DH * 2);
  __bf16* Vtb   = (__bf16*)carve((size_t)BDIM * NH * DH * NSEQ * 2);
  float*  biasb = (float*)carve((size_t)BDIM * NH * NSEQ * NSEQ * 4);
  __bf16* obf   = (__bf16*)carve((size_t)BDIM * NSEQ * INNER * 2);

  const size_t gemm_lds = (size_t)64 * BS_STRIDE * 2;   // 66,560 B

  k_node_ln<<<BDIM * NSEQ / 8, 256, 0, stream>>>(node, nnw, nnb, xln);
  k_transpose<<<(DMODEL * 3 * INNER + 255) / 256, 256, 0, stream>>>(qkvw, wqkvT, DMODEL, 3 * INNER);
  k_transpose<<<(DMODEL * INNER + 255) / 256, 256, 0, stream>>>(gw, wgT, DMODEL, INNER);
  k_transpose<<<(INNER * DMODEL + 255) / 256, 256, 0, stream>>>(ow, woutT, INNER, DMODEL);
  k_transpose<<<(PD * NH + 255) / 256, 256, 0, stream>>>(bw, bwT, PD, NH);

  k_gemm<<<(BDIM * NSEQ / 128) * (3 * INNER / 64), 256, gemm_lds, stream>>>(
      xln, wqkvT, qkvb, qkvf, 3 * INNER);
  k_gemm<<<(BDIM * NSEQ / 128) * (INNER / 64), 256, gemm_lds, stream>>>(
      xln, wgT, gb, gf, INNER);
  k_qkv_post<<<BDIM * NSEQ / 8, 256, 0, stream>>>(qkvf, qlnw, qlnb, klnw, klnb, Qb, Kb, Vtb);
  k_pair_bias<<<BDIM * NSEQ * (NSEQ / 16) / 8, 256, 0, stream>>>(pair, mask, pnw, pnb, bwT, biasb);
  k_attn<<<BDIM * NH * (NSEQ / 16) / 8, 256, 0, stream>>>(Qb, Kb, Vtb, biasb, gf, obf);
  k_gemm<<<(BDIM * NSEQ / 128) * (DMODEL / 64), 256, gemm_lds, stream>>>(
      obf, woutT, ob, (float*)d_out, DMODEL);
}
